// DecoderRNN_86586540687481
// MI455X (gfx1250) — matrix operand order repaired and checked by reading, not compiled, not run on hardware
//
#include <hip/hip_runtime.h>
#include <hip/hip_bf16.h>
#include <stddef.h>
#include <stdint.h>

typedef __attribute__((ext_vector_type(16))) __bf16 v16bf;
typedef __attribute__((ext_vector_type(8)))  __bf16 v8bf;
typedef __attribute__((ext_vector_type(8)))  float  v8f;

#define B_   64
#define T_   32
#define E_   512
#define H_   512
#define V_   10000
#define G4_  2048      // 4*H
#define NSTEP_ 33      // 1 feature step + 32 caption steps
#define PADH 40        // LDS row pitch in halfs (80B) to spread b128 reads across banks

union BFrag16 { v16bf v; v8bf h[2]; };

// A-matrix fragment (16x32 bf16, MxK), rowPtr = base of this lane's M-row.
// Lanes 0-15: K {0..7, 16..23}; lanes 16-31: K {8..15, 24..31}  (ksel = lane>>4)
__device__ __forceinline__ v16bf load_a_frag(const __bf16* rowPtr, int kk, int ksel) {
    BFrag16 a;
    a.h[0] = *(const v8bf*)(rowPtr + kk + ksel * 8);
    a.h[1] = *(const v8bf*)(rowPtr + kk + 16 + ksel * 8);
    return a.v;
}

// B-matrix fragment (32x16 bf16, KxN) from an LDS tile [16 rows][PADH halfs].
// B-column n = lane&15; lanes 0-15: K {0..7, 16..23}; lanes 16-31: K {8..15, 24..31}.
__device__ __forceinline__ v16bf load_b_lds(const __bf16* tileBase, int n, int ksel) {
    BFrag16 b;
    b.h[0] = *(const v8bf*)(tileBase + n * PADH + ksel * 8);
    b.h[1] = *(const v8bf*)(tileBase + n * PADH + 16 + ksel * 8);
    return b.v;
}

// ---- CDNA5 async global->LDS copy (ASYNCcnt-tracked, no VGPR round trip) ----
// Generic pointer to a __shared__ object carries the LDS byte offset in its
// low 32 bits.
__device__ __forceinline__ unsigned lds_off(const void* p) {
    return (unsigned)(uintptr_t)p;
}
__device__ __forceinline__ void async_copy_b128(unsigned ldsAddr, const void* gaddr) {
    asm volatile("global_load_async_to_lds_b128 %0, %1, off"
                 :: "v"(ldsAddr), "v"(gaddr) : "memory");
}
__device__ __forceinline__ void wait_async_le0() {
    asm volatile("s_wait_asynccnt 0x0" ::: "memory");
}
__device__ __forceinline__ void wait_async_le1() {
    asm volatile("s_wait_asynccnt 0x1" ::: "memory");
}
__device__ __forceinline__ void wait_async_le2() {
    asm volatile("s_wait_asynccnt 0x2" ::: "memory");
}

__device__ __forceinline__ float sigmoid_f(float x) {
    return 1.0f / (1.0f + __expf(-x));
}
__device__ __forceinline__ float tanh_f(float x) {
    return 2.0f / (1.0f + __expf(-2.0f * x)) - 1.0f;
}

// ---------------------------------------------------------------- utilities
__global__ void f32_to_bf16_kernel(const float* __restrict__ src,
                                   __bf16* __restrict__ dst, int n) {
    int i = blockIdx.x * blockDim.x + threadIdx.x;
    int stride = gridDim.x * blockDim.x;
    for (; i < n; i += stride) dst[i] = (__bf16)src[i];
}

// x_all (NSTEP_, B, E) bf16: slot 0 = image features, slot 1+t = embed_W[captions[:,t]]
__global__ void prep_x_kernel(const float* __restrict__ features,
                              const int*   __restrict__ captions,
                              const float* __restrict__ embedW,
                              __bf16* __restrict__ x_all) {
    int row = blockIdx.x;              // 0 .. NSTEP_*B_-1
    int s = row >> 6;
    int b = row & 63;
    const float* src;
    if (s == 0) {
        src = features + (size_t)b * E_;
    } else {
        int tok = captions[b * T_ + (s - 1)];
        src = embedW + (size_t)tok * E_;
    }
    __bf16* dst = x_all + (size_t)row * E_;
    for (int e = threadIdx.x; e < E_; e += blockDim.x) dst[e] = (__bf16)src[e];
}

__global__ void zero_state_kernel(__bf16* __restrict__ h0, float* __restrict__ c) {
    int i = blockIdx.x * blockDim.x + threadIdx.x;   // 0 .. B_*H_-1
    h0[i] = (__bf16)0.0f;
    c[i]  = 0.0f;
}

// ---------------------------------------------------------------- fused LSTM step
// Block = 128 threads = 4 waves. blockIdx.x -> j0 = hidden-column tile (32 blocks).
// Wave w = M-tile; its 4 accumulators are the 4 gates (i,f,g,o) for that (m,j)
// patch, so the cell update happens in registers. W_ih/W_hh tiles stream through
// LDS via async copies, triple-buffered with prefetch distance 2.
__global__ __launch_bounds__(128)
void lstm_step_kernel(const __bf16* __restrict__ x,      // (64,512)
                      const __bf16* __restrict__ hprev,  // (64,512)
                      const __bf16* __restrict__ Wih,    // (2048,512)
                      const __bf16* __restrict__ Whh,    // (2048,512)
                      const float*  __restrict__ b_ih,   // (2048)
                      const float*  __restrict__ b_hh,   // (2048)
                      float*  __restrict__ c,            // (64,512) in/out
                      __bf16* __restrict__ h_next) {     // (64,512)
    __shared__ __bf16 lds[3][4][16][PADH];               // 3 bufs x 4 gates x 16 rows x 40h

    const int tid  = threadIdx.x;
    const int lane = tid & 31;
    const int wave = tid >> 5;          // M-tile
    const int mrow = lane & 15;
    const int ksel = lane >> 4;
    const int j0   = blockIdx.x * 16;
    const int m0   = wave * 16;

    const __bf16* axrow = x     + (size_t)(m0 + mrow) * E_;
    const __bf16* ahrow = hprev + (size_t)(m0 + mrow) * H_;

    // 256 b128 slots per K-step; this thread owns slots tid and tid+128.
    // slot s: nt = s>>6 (gate), n = s&15 (B column), kq = (s&63)>>4 (8-half chunk)
    const int s0 = tid,        s0_nt = s0 >> 6, s0_n = s0 & 15, s0_kq = (s0 & 63) >> 4;
    const int s1 = tid + 128,  s1_nt = s1 >> 6, s1_n = s1 & 15, s1_kq = (s1 & 63) >> 4;
    const size_t roff0 = (size_t)(s0_nt * H_ + j0 + s0_n) * E_ + s0_kq * 8;
    const size_t roff1 = (size_t)(s1_nt * H_ + j0 + s1_n) * E_ + s1_kq * 8;

    unsigned la0[3], la1[3];
#pragma unroll
    for (int bfi = 0; bfi < 3; ++bfi) {
        la0[bfi] = lds_off(&lds[bfi][s0_nt][s0_n][s0_kq * 8]);
        la1[bfi] = lds_off(&lds[bfi][s1_nt][s1_n][s1_kq * 8]);
    }

    auto stage = [&](int iterIdx, int buf) {
        const __bf16* W = (iterIdx < 16) ? Wih : Whh;
        const int kk = (iterIdx & 15) * 32;
        async_copy_b128(la0[buf], W + roff0 + kk);
        async_copy_b128(la1[buf], W + roff1 + kk);
    };

    v8f acc[4] = {};

    // Compute body: load all 4 B frags into distinct regs (batched ds_loads,
    // one partial wait), then 4 back-to-back WMMAs on independent accumulators.
    auto body = [&](int it) {
        const __bf16* arow = (it < 16) ? axrow : ahrow;
        const int kk = (it & 15) * 32;
        v16bf a = load_a_frag(arow, kk, ksel);
        const __bf16* base = &lds[it % 3][0][0][0];
        v16bf bfr[4];
#pragma unroll
        for (int g = 0; g < 4; ++g)
            bfr[g] = load_b_lds(base + g * (16 * PADH), mrow, ksel);
#pragma unroll
        for (int g = 0; g < 4; ++g)
            acc[g] = __builtin_amdgcn_wmma_f32_16x16x32_bf16(false, a, false, bfr[g],
                                                             (short)0, acc[g], false, false);
    };

    stage(0, 0);
    stage(1, 1);
    wait_async_le2();                   // iter 0's two copies done; iter 1's may fly
    __syncthreads();
#pragma unroll 1
    for (int it = 0; it < 30; ++it) {   // steady state: unconditional prefetch + wait
        stage(it + 2, (it + 2) % 3);
        body(it);
        wait_async_le2();               // stage(it+1) done; stage(it+2) stays in flight
        __syncthreads();
    }
    body(30);
    wait_async_le0();                   // drain stage(31)
    __syncthreads();
    body(31);

    // fused pointwise cell update (gate order i,f,g,o)
    const int j = j0 + mrow;
    const float sb0 = b_ih[j]           + b_hh[j];
    const float sb1 = b_ih[j + H_]      + b_hh[j + H_];
    const float sb2 = b_ih[j + 2 * H_]  + b_hh[j + 2 * H_];
    const float sb3 = b_ih[j + 3 * H_]  + b_hh[j + 3 * H_];
#pragma unroll
    for (int r = 0; r < 8; ++r) {
        int m = m0 + ksel * 8 + r;
        size_t ci = (size_t)m * H_ + j;
        float i_ = sigmoid_f(acc[0][r] + sb0);
        float f_ = sigmoid_f(acc[1][r] + sb1);
        float g_ = tanh_f   (acc[2][r] + sb2);
        float o_ = sigmoid_f(acc[3][r] + sb3);
        float cn = f_ * c[ci] + i_ * g_;
        c[ci] = cn;
        h_next[ci] = (__bf16)(o_ * tanh_f(cn));
    }
}

// ---------------------------------------------------------------- FC projection GEMM
// out(b,t,v) = Hseq(2048x512) @ Wfc^T + bfc.
// Block = 256 threads = 8 waves; block tile = 256 rows x 64 cols; wave = 2x4 tiles.
// B tile (4KB/K-step) streams through LDS via async copies, triple-buffered.
__global__ __launch_bounds__(256)
void fc_kernel(const __bf16* __restrict__ Hseq,  // (T*B, H), row m = t*64+b
               const __bf16* __restrict__ Wfc,   // (10000, 512)
               const float*  __restrict__ bfc,   // (10000)
               float* __restrict__ out) {        // (B, T, V)
    __shared__ __bf16 lds[3][4][16][PADH];

    const int tid   = threadIdx.x;
    const int lane  = tid & 31;
    const int wave  = tid >> 5;                  // 0..7
    const int mrow  = lane & 15;
    const int ksel  = lane >> 4;
    const int n0    = blockIdx.x * 64;
    const int rowbase = blockIdx.y * 256 + wave * 32;

    const __bf16* arow0 = Hseq + (size_t)(rowbase + mrow) * H_;
    const __bf16* arow1 = arow0 + (size_t)16 * H_;

    // staging: one b128 slot per thread per K-step
    const int s_nt = tid >> 6;
    const int s_n  = tid & 15;
    const int s_kq = (tid & 63) >> 4;
    int s_row = n0 + s_nt * 16 + s_n;
    if (s_row >= V_) s_row = V_ - 1;             // clamp OOB rows (tiles >= 625 never stored)
    const __bf16* s_src = Wfc + (size_t)s_row * H_ + s_kq * 8;

    unsigned la[3];
#pragma unroll
    for (int bfi = 0; bfi < 3; ++bfi)
        la[bfi] = lds_off(&lds[bfi][s_nt][s_n][s_kq * 8]);

    v8f acc[2][4] = {};

    auto body = [&](int it) {
        const int kk = it * 32;
        v16bf a0 = load_a_frag(arow0, kk, ksel);
        v16bf a1 = load_a_frag(arow1, kk, ksel);
        const __bf16* base = &lds[it % 3][0][0][0];
        v16bf bfr[4];
#pragma unroll
        for (int nt = 0; nt < 4; ++nt)
            bfr[nt] = load_b_lds(base + nt * (16 * PADH), mrow, ksel);
#pragma unroll
        for (int nt = 0; nt < 4; ++nt) {
            acc[0][nt] = __builtin_amdgcn_wmma_f32_16x16x32_bf16(false, a0, false, bfr[nt],
                                                                 (short)0, acc[0][nt], false, false);
            acc[1][nt] = __builtin_amdgcn_wmma_f32_16x16x32_bf16(false, a1, false, bfr[nt],
                                                                 (short)0, acc[1][nt], false, false);
        }
    };

    async_copy_b128(la[0], s_src + 0);
    async_copy_b128(la[1], s_src + 32);
    wait_async_le1();                   // iter 0's copy done; iter 1's may fly
    __syncthreads();
#pragma unroll 1
    for (int it = 0; it < 14; ++it) {   // steady state
        async_copy_b128(la[(it + 2) % 3], s_src + (it + 2) * 32);
        body(it);
        wait_async_le1();
        __syncthreads();
    }
    body(14);
    wait_async_le0();                   // drain stage(15)
    __syncthreads();
    body(15);

#pragma unroll
    for (int mt = 0; mt < 2; ++mt) {
#pragma unroll
        for (int nt = 0; nt < 4; ++nt) {
            int jc = n0 + nt * 16;
            if (jc < V_) {                       // uniform per wave -> scalar branch
                int col = jc + mrow;
                float bias = bfc[col];
#pragma unroll
                for (int r = 0; r < 8; ++r) {
                    int m = rowbase + mt * 16 + ksel * 8 + r;
                    int t = m >> 6;
                    int b = m & 63;
                    out[((size_t)b * T_ + t) * V_ + col] = acc[mt][nt][r] + bias;
                }
            }
        }
    }
}

// ---------------------------------------------------------------- launch
extern "C" void kernel_launch(void* const* d_in, const int* in_sizes, int n_in,
                              void* d_out, int out_size, void* d_ws, size_t ws_size,
                              hipStream_t stream) {
    (void)in_sizes; (void)n_in; (void)out_size; (void)ws_size;

    const float* features = (const float*)d_in[0];   // (B,E)
    const int*   captions = (const int*)  d_in[1];   // (B,T)
    const float* embedW   = (const float*)d_in[2];   // (V,E)
    const float* Wih_f    = (const float*)d_in[3];   // (4H,E)
    const float* Whh_f    = (const float*)d_in[4];   // (4H,H)
    const float* b_ih     = (const float*)d_in[5];   // (4H)
    const float* b_hh     = (const float*)d_in[6];   // (4H)
    const float* Wfc_f    = (const float*)d_in[7];   // (V,H)
    const float* bfc      = (const float*)d_in[8];   // (V)
    float* out = (float*)d_out;

    char* ws = (char*)d_ws;
    size_t off = 0;
    auto carve = [&](size_t bytes) -> void* {
        void* p = ws + off;
        off = (off + bytes + 255) & ~(size_t)255;
        return p;
    };
    __bf16* Wih     = (__bf16*)carve((size_t)G4_ * E_ * 2);
    __bf16* Whh     = (__bf16*)carve((size_t)G4_ * H_ * 2);
    __bf16* Wfc     = (__bf16*)carve((size_t)V_  * H_ * 2);
    __bf16* x_all   = (__bf16*)carve((size_t)NSTEP_ * B_ * E_ * 2);
    __bf16* h_state = (__bf16*)carve((size_t)(NSTEP_ + 1) * B_ * H_ * 2);
    float*  c_state = (float*) carve((size_t)B_ * H_ * 4);

    // 1) weights f32 -> bf16 (L2-resident thereafter)
    {
        int n1 = G4_ * E_;
        f32_to_bf16_kernel<<<(n1 + 255) / 256, 256, 0, stream>>>(Wih_f, Wih, n1);
        f32_to_bf16_kernel<<<(n1 + 255) / 256, 256, 0, stream>>>(Whh_f, Whh, n1);
        int n2 = V_ * H_;
        f32_to_bf16_kernel<<<(n2 + 255) / 256, 256, 0, stream>>>(Wfc_f, Wfc, n2);
    }

    // 2) features + embedding gather -> bf16 (NSTEP_, B, E)
    prep_x_kernel<<<NSTEP_ * B_, 128, 0, stream>>>(features, captions, embedW, x_all);

    // 3) zero h0 / c
    zero_state_kernel<<<(B_ * H_) / 256, 256, 0, stream>>>(h_state, c_state);

    // 4) 33 serial fused LSTM steps (GEMM + in-register cell update)
    for (int s = 0; s < NSTEP_; ++s) {
        lstm_step_kernel<<<H_ / 16, 128, 0, stream>>>(
            x_all   + (size_t)s * B_ * E_,
            h_state + (size_t)s * B_ * H_,
            Wih, Whh, b_ih, b_hh,
            c_state,
            h_state + (size_t)(s + 1) * B_ * H_);
    }

    // 5) one big FC GEMM over all caption steps (rows m = t*64+b -> h after step t+1)
    {
        dim3 grid((V_ / 16 + 3) / 4, (T_ * B_) / 256);   // (157, 8)
        fc_kernel<<<grid, 256, 0, stream>>>(
            h_state + (size_t)2 * B_ * H_,               // slot 2 == h after caption step 0
            Wfc, bfc, out);
    }
}
